// SK_CLSTM_27616639713768
// MI455X (gfx1250) — matrix operand order repaired and checked by reading, not compiled, not run on hardware
//
#include <hip/hip_runtime.h>
#include <stddef.h>

// ---------------- problem constants ----------------
#define BQ    128            // batch
#define TT    12             // time steps
#define CIN   10             // input channels
#define HID   64             // hidden channels
#define HW    25             // spatial H = W
#define SPAT  (HW*HW)        // 625
#define CTOT  (CIN+HID)      // 74 concat channels
#define NGATE (4*HID)        // 256 gate channels
#define HP    27             // padded spatial (halo of 1)
#define PIX   (HP*HP)        // 729
#define CP    96             // channels padded to 3 chunks of 32
#define CPU   (CP/2)         // 48 u32 per pixel
#define PIN_U32 (BQ*PIX*CPU)     // 4,478,976 u32 (17.9 MB)
#define WPK_U32 (NGATE*9*CPU)    // 110,592 u32
#define MTOT  (BQ*SPAT)      // 80000, divisible by 64

typedef __attribute__((ext_vector_type(16))) __bf16        v16bf;
typedef __attribute__((ext_vector_type(8)))  float         v8f;
typedef __attribute__((ext_vector_type(8)))  unsigned int  v8u;

__device__ __forceinline__ unsigned short f32_to_bf16(float f) {
    unsigned int u = __builtin_bit_cast(unsigned int, f);
    u += 0x7FFFu + ((u >> 16) & 1u);          // round-to-nearest-even
    return (unsigned short)(u >> 16);
}

// =====================================================================
// Implicit-GEMM conv, operands pre-packed in bf16:
//   A: pin[b][iy27][ix27][cp96]  (zero halo + zero pad channels)
//   B: wpk[n][tap9][cp96]
// K order: k = (tap, chunk*32 + c). 27 K-steps of 32.
// Block: 256 thr = 8 waves; tile 64M x 64N; wave (mi,ni) owns M rows
// {mi*16, mi*16+32} x N cols ni*16 -> two accumulators share one B frag.
// grid = (1250, 4); blockIdx.y = gate group (i,f,o,g).
// =====================================================================
__global__ __launch_bounds__(256)
void convlstm_gates_wmma(const unsigned int* __restrict__ pin,
                         const unsigned int* __restrict__ wpk,
                         float* __restrict__ gates)       // (B,256,25,25) f32
{
    __shared__ __align__(16) unsigned int lds_a[64 * 16];   // 64 M x 32 K bf16, 4KB
    __shared__ __align__(16) unsigned int lds_b[64 * 16];   // 64 N x 32 K bf16, 4KB

    const int tid  = threadIdx.x;
    const int lane = tid & 31;
    const int wave = tid >> 5;
    const int mi   = wave >> 2;               // 0..1
    const int ni   = wave & 3;                // 0..3
    const int nb   = blockIdx.y;              // gate group
    const int m_base = blockIdx.x * 64;

    // ---- loop-invariant staging addresses (one row of 4 u32 per thread) ----
    const int row = tid >> 2;                 // 0..63 tile row
    const int u0  = (tid & 3) * 4;            // u32 column within row

    // A source: top-left of this output position's 3x3 window in padded coords
    int m  = m_base + row;
    int b0 = m / SPAT;
    int p0 = m - b0 * SPAT;
    int y0 = p0 / HW, x0 = p0 - y0 * HW;
    const size_t apix = ((size_t)b0 * PIX + (size_t)y0 * HP + x0) * CPU;

    // B source: row n of the repacked weights
    const int nrow_g = nb * 64 + row;
    const size_t wbase = (size_t)nrow_g * 9 * CPU;

    __builtin_prefetch(wpk + (size_t)(nb * 64) * 9 * CPU + (size_t)tid * 16, 0, 1);

    v8f acc0 = {};
    v8f acc1 = {};

    // fragment gather constants (CDNA5 wave32 WMMA bf16 layouts)
    const int mrow0 = (mi * 16 + (lane & 15)) * 16;
    const int mrow1 = (mi * 16 + 32 + (lane & 15)) * 16;
    const int nrow  = (ni * 16 + (lane & 15)) * 16;
    const int khalf = lane >> 4;

    for (int tap = 0; tap < 9; ++tap) {
        const int ky = tap / 3, kx = tap - ky * 3;
        const int taps_off = (ky * HP + kx) * CPU;         // scalar
        for (int chunk = 0; chunk < 3; ++chunk) {
            const int c_off = chunk * 16;                  // u32 offset of K-chunk

            // ---- stage A: one b128 load + one b128 LDS store per thread ----
            uint4 av = *(const uint4*)(pin + apix + (size_t)(taps_off + c_off + u0));
            *(uint4*)&lds_a[row * 16 + u0] = av;

            // ---- stage B: one b128 load + one b128 LDS store per thread ----
            uint4 bv = *(const uint4*)(wpk + wbase + (size_t)(tap * CPU + c_off + u0));
            *(uint4*)&lds_b[row * 16 + u0] = bv;

            __syncthreads();

            // A 16x32 bf16: lane->M=lane&15; VGPR v: k0=(v>>2)*16+(lane>>4)*8+(v&3)*2
            // B 32x16 bf16: lane->N=lane&15; VGPR v: k0=(v>>2)*16+(lane>>4)*8+(v&3)*2
            v8u au0, au1, bu;
            #pragma unroll
            for (int vv = 0; vv < 8; ++vv) {
                int kpA = (vv >> 2) * 8 + khalf * 4 + (vv & 3);
                int kpB = (vv >> 2) * 8 + khalf * 4 + (vv & 3);
                au0[vv] = lds_a[mrow0 + kpA];
                au1[vv] = lds_a[mrow1 + kpA];
                bu [vv] = lds_b[nrow  + kpB];
            }
            v16bf af0 = __builtin_bit_cast(v16bf, au0);
            v16bf af1 = __builtin_bit_cast(v16bf, au1);
            v16bf bf  = __builtin_bit_cast(v16bf, bu);

            acc0 = __builtin_amdgcn_wmma_f32_16x16x32_bf16(
                       false, af0, false, bf, (short)0, acc0, false, false);
            acc1 = __builtin_amdgcn_wmma_f32_16x16x32_bf16(
                       false, af1, false, bf, (short)0, acc1, false, false);

            __syncthreads();
        }
    }

    // ---- write D: VGPR r -> M = r + (lane>>4)*8, N = lane&15 ----
    const int nc = nb * 64 + ni * 16 + (lane & 15);
    #pragma unroll
    for (int r = 0; r < 8; ++r) {
        int mm = m_base + mi * 16 + r + khalf * 8;
        int b  = mm / SPAT;
        int pos = mm - b * SPAT;
        gates[((size_t)b * NGATE + nc) * SPAT + pos] = acc0[r];
        int mm1 = mm + 32;
        b   = mm1 / SPAT;
        pos = mm1 - b * SPAT;
        gates[((size_t)b * NGATE + nc) * SPAT + pos] = acc1[r];
    }
}

// =====================================================================
// Pack x_t channels (0..9) into pin interior (halo/pad stay zero).
// One u32 (2 channels) per thread: idx = (b*625 + pos)*5 + j.
// =====================================================================
__global__ void pack_x(const float* __restrict__ x, int t,
                       unsigned int* __restrict__ pin)
{
    int idx = blockIdx.x * blockDim.x + threadIdx.x;
    if (idx >= BQ * SPAT * (CIN / 2)) return;
    int j   = idx % (CIN / 2);
    int r   = idx / (CIN / 2);
    int pos = r % SPAT;
    int b   = r / SPAT;
    int y = pos / HW, xq = pos - y * HW;
    const float* xp = x + (((size_t)b * TT + t) * CIN + 2 * j) * SPAT + pos;
    unsigned int pr = (unsigned int)f32_to_bf16(xp[0])
                    | ((unsigned int)f32_to_bf16(xp[SPAT]) << 16);
    pin[((size_t)b * PIX + (size_t)(y + 1) * HP + (xq + 1)) * CPU + j] = pr;
}

// =====================================================================
// Repack conv weights: (256,74,3,3) f32 -> wpk[n][tap][cp96] bf16.
// =====================================================================
__global__ void pack_w(const float* __restrict__ w,
                       unsigned int* __restrict__ wpk)
{
    int idx = blockIdx.x * blockDim.x + threadIdx.x;
    if (idx >= WPK_U32) return;
    int cpair = idx % CPU;
    int r     = idx / CPU;
    int tap   = r % 9;
    int n     = r / 9;
    unsigned int pr = 0;
    #pragma unroll
    for (int e = 0; e < 2; ++e) {
        int c = cpair * 2 + e;
        float v = (c < CTOT) ? w[(size_t)(n * CTOT + c) * 9 + tap] : 0.0f;
        pr |= (unsigned int)f32_to_bf16(v) << (16 * e);
    }
    wpk[idx] = pr;
}

// =====================================================================
// Fused LSTM cell update; also streams bf16(h_new) into pin for step t+1.
// =====================================================================
__global__ void lstm_update(const float* __restrict__ gates,
                            const float* __restrict__ bias,
                            float* __restrict__ h, float* __restrict__ c,
                            unsigned short* __restrict__ pin16)
{
    int idx = blockIdx.x * blockDim.x + threadIdx.x;       // over B*HID*SPAT
    if (idx >= BQ * HID * SPAT) return;
    int pos  = idx % SPAT;
    int tmpq = idx / SPAT;
    int hid  = tmpq % HID;
    int b    = tmpq / HID;
    size_t base = (size_t)b * NGATE * SPAT;
    float gi = gates[base + (size_t)(0 * HID + hid) * SPAT + pos] + bias[0 * HID + hid];
    float gf = gates[base + (size_t)(1 * HID + hid) * SPAT + pos] + bias[1 * HID + hid];
    float go = gates[base + (size_t)(2 * HID + hid) * SPAT + pos] + bias[2 * HID + hid];
    float gg = gates[base + (size_t)(3 * HID + hid) * SPAT + pos] + bias[3 * HID + hid];
    float ig = 1.0f / (1.0f + __expf(-gi));
    float fg = 1.0f / (1.0f + __expf(-gf));
    float og = 1.0f / (1.0f + __expf(-go));
    float gv = tanhf(gg);
    float cn = fg * c[idx] + ig * gv;
    float hn = og * tanhf(cn);
    c[idx] = cn;
    h[idx] = hn;
    int y = pos / HW, xq = pos - y * HW;
    pin16[((size_t)b * PIX + (size_t)(y + 1) * HP + (xq + 1)) * CP + CIN + hid]
        = f32_to_bf16(hn);
}

// =====================================================================
// MLP head: relu(h) flatten (40000) -> 16 -> 8 -> 2 -> log_softmax.
// =====================================================================
__global__ __launch_bounds__(256)
void mlp_head(const float* __restrict__ h,
              const float* __restrict__ w1, const float* __restrict__ b1,
              const float* __restrict__ w2, const float* __restrict__ b2,
              const float* __restrict__ w3, const float* __restrict__ b3,
              float* __restrict__ out)
{
    __shared__ float sred[256];
    __shared__ float z1[16];
    const int b   = blockIdx.x;
    const int tid = threadIdx.x;
    const float* hb = h + (size_t)b * (HID * SPAT);

    float acc[16];
    #pragma unroll
    for (int o = 0; o < 16; ++o) acc[o] = 0.0f;

    for (int k = tid; k < HID * SPAT; k += 256) {
        float hv = hb[k];
        hv = hv > 0.0f ? hv : 0.0f;
        const float* wr = w1 + (size_t)k * 16;
        #pragma unroll
        for (int o = 0; o < 16; ++o) acc[o] += hv * wr[o];
    }
    for (int o = 0; o < 16; ++o) {
        sred[tid] = acc[o];
        __syncthreads();
        for (int s = 128; s > 0; s >>= 1) {
            if (tid < s) sred[tid] += sred[tid + s];
            __syncthreads();
        }
        if (tid == 0) { float z = sred[0] + b1[o]; z1[o] = z > 0.0f ? z : 0.0f; }
        __syncthreads();
    }
    if (tid == 0) {
        float z2[8];
        #pragma unroll
        for (int j = 0; j < 8; ++j) {
            float s = b2[j];
            #pragma unroll
            for (int o = 0; o < 16; ++o) s += z1[o] * w2[o * 8 + j];
            z2[j] = s > 0.0f ? s : 0.0f;
        }
        float l0 = b3[0], l1 = b3[1];
        #pragma unroll
        for (int j = 0; j < 8; ++j) { l0 += z2[j] * w3[j * 2 + 0]; l1 += z2[j] * w3[j * 2 + 1]; }
        float mx  = l0 > l1 ? l0 : l1;
        float lse = mx + __logf(__expf(l0 - mx) + __expf(l1 - mx));
        out[b * 2 + 0] = l0 - lse;
        out[b * 2 + 1] = l1 - lse;
    }
}

__global__ void zero_u32(unsigned int* __restrict__ p, int n)
{
    int i = blockIdx.x * blockDim.x + threadIdx.x;
    if (i < n) p[i] = 0u;
}

// =====================================================================
// Host side
// =====================================================================
extern "C" void kernel_launch(void* const* d_in, const int* in_sizes, int n_in,
                              void* d_out, int out_size, void* d_ws, size_t ws_size,
                              hipStream_t stream)
{
    (void)in_sizes; (void)n_in; (void)out_size; (void)ws_size;
    const float* x      = (const float*)d_in[0];
    const float* conv_w = (const float*)d_in[1];
    const float* conv_b = (const float*)d_in[2];
    const float* w1     = (const float*)d_in[3];
    const float* b1     = (const float*)d_in[4];
    const float* w2     = (const float*)d_in[5];
    const float* b2     = (const float*)d_in[6];
    const float* w3     = (const float*)d_in[7];
    const float* b3     = (const float*)d_in[8];
    float* out = (float*)d_out;

    // ws layout: gates f32 | h f32 | c f32 | pin bf16 | wpk bf16
    float* gates = (float*)d_ws;                            // 20,480,000 f32
    float* hbuf  = gates + (size_t)BQ * NGATE * SPAT;       //  5,120,000 f32
    float* cbuf  = hbuf  + (size_t)BQ * HID * SPAT;         //  5,120,000 f32
    unsigned int* pin = (unsigned int*)(cbuf + (size_t)BQ * HID * SPAT);
    unsigned int* wpk = pin + PIN_U32;
    unsigned short* pin16 = (unsigned short*)pin;

    // zero h, c and pin (h/c/pin are contiguous): halo + pad channels must be 0
    const int zero_dw = 2 * BQ * HID * SPAT + PIN_U32;      // h,c as dwords + pin
    zero_u32<<<(zero_dw + 255) / 256, 256, 0, stream>>>((unsigned int*)hbuf, zero_dw);

    pack_w<<<(WPK_U32 + 255) / 256, 256, 0, stream>>>(conv_w, wpk);

    dim3 conv_grid(MTOT / 64, NGATE / 64);                  // (1250, 4)
    const int px_threads = BQ * SPAT * (CIN / 2);           // 400,000
    const int upd_elems  = BQ * HID * SPAT;                 // 5,120,000

    for (int t = 0; t < TT; ++t) {
        pack_x<<<(px_threads + 255) / 256, 256, 0, stream>>>(x, t, pin);
        convlstm_gates_wmma<<<conv_grid, 256, 0, stream>>>(pin, wpk, gates);
        lstm_update<<<(upd_elems + 255) / 256, 256, 0, stream>>>(
            gates, conv_b, hbuf, cbuf, pin16);
    }

    mlp_head<<<BQ, 256, 0, stream>>>(hbuf, w1, b1, w2, b2, w3, b3, out);
}
